// SwinTransformerBlock_29068338660126
// MI455X (gfx1250) — matrix operand order repaired and checked by reading, not compiled, not run on hardware
//
#include <hip/hip_runtime.h>
#include <hip/hip_bf16.h>

typedef _Float16 h16;
typedef __attribute__((ext_vector_type(16))) _Float16 v16h;
typedef __attribute__((ext_vector_type(8)))  float    v8f;
typedef unsigned int u32;
typedef unsigned long long u64;

// ---------------------------------------------------------------------------
// WMMA helpers (gfx1250: D = A(16x32 f16) x B(32x16 f16) + C(16x16 f32))
// ---------------------------------------------------------------------------
__device__ inline v8f wmma16(v16h a, v16h b, v8f c) {
  return __builtin_amdgcn_wmma_f32_16x16x32_f16(false, a, false, b, (short)0, c,
                                                false, false);
}

// A fragment 16x32: source row-major M[m][k], leading dim ld (elements).
__device__ inline v16h ldfragA(const h16* p0, int ld) {
  int lane = threadIdx.x & 31;
  int m = lane & 15, hi = lane >> 4;
  const h16* p = p0 + m * ld + hi * 8;
  v16h f;
  u32* u = (u32*)&f;
#pragma unroll
  for (int j = 0; j < 4; ++j) {
    u[j]     = *(const u32*)(p + 2 * j);
    u[j + 4] = *(const u32*)(p + 16 + 2 * j);
  }
  return f;
}

// B fragment 32x16 where B[k][n] = M[n][k], M row-major with leading dim ld.
__device__ inline v16h ldfragBT(const h16* p0, int ld) {
  int lane = threadIdx.x & 31;
  int n = lane & 15, hi = lane >> 4;
  const h16* p = p0 + n * ld + hi * 8;
  v16h f;
  u32* u = (u32*)&f;
#pragma unroll
  for (int j = 0; j < 4; ++j) {
    u[j]     = *(const u32*)(p + 2 * j);
    u[j + 4] = *(const u32*)(p + 16 + 2 * j);
  }
  return f;
}

// Store D (16x16 f32) to row-major dst with leading dim ld.
__device__ inline void stD(float* p0, int ld, v8f d) {
  int lane = threadIdx.x & 31;
  int n = lane & 15, hi = lane >> 4;
#pragma unroll
  for (int r = 0; r < 8; ++r) p0[(r + 8 * hi) * ld + n] = d[r];
}

// ---------------------------------------------------------------------------
// gfx1250 async global->LDS copy (16 B per lane, ASYNCcnt-tracked)
// ---------------------------------------------------------------------------
__device__ inline void async_cp16(u32 lds_off, u64 gaddr) {
  asm volatile("global_load_async_to_lds_b128 %0, %1, off"
               :: "v"(lds_off), "v"(gaddr) : "memory");
}
__device__ inline void wait_async0() {
  asm volatile("s_wait_asynccnt 0x0" ::: "memory");
}

// ---------------------------------------------------------------------------
// Shift-window mask region (SwinV2, H=W=64, WS=8, SHIFT=4)
// ---------------------------------------------------------------------------
__device__ inline int regionOf(int widx, int tok) {
  int wh = widx >> 3, ww = widx & 7;
  int r = tok >> 3, c = tok & 7;
  int h = wh * 8 + r, w = ww * 8 + c;
  int rh = (h < 56) ? 0 : (h < 60 ? 1 : 2);
  int rw = (w < 56) ? 0 : (w < 60 ? 1 : 2);
  return rh * 3 + rw;
}

// ---------------------------------------------------------------------------
// K0: fp32 -> f16 transpose of weights: dst[n*K + k] = src[k*N + n]
// ---------------------------------------------------------------------------
__global__ void k_transpose_h16(const float* __restrict__ src,
                                h16* __restrict__ dst, int K, int N) {
  int idx = blockIdx.x * 256 + threadIdx.x;
  if (idx < K * N) {
    int n = idx / K, k = idx % K;
    dst[idx] = (h16)src[k * N + n];
  }
}

// K0b: fused qkv bias vector [q_bias | 0 | v_bias]
__global__ void k_qkvbias(const float* __restrict__ qb,
                          const float* __restrict__ vb,
                          float* __restrict__ dst) {
  int i = blockIdx.x * 256 + threadIdx.x;
  if (i < 768) dst[i] = (i < 256) ? qb[i] : ((i < 512) ? 0.f : vb[i - 512]);
}

// ---------------------------------------------------------------------------
// K1a: CPB MLP table:  relu(RCT @ w1 + b1) @ w2   -> tab[225][8]
// ---------------------------------------------------------------------------
__device__ inline float rctv(int i) {
  float d = (float)(i - 7);
  float t = d * (8.0f / 7.0f);
  float v = log2f(fabsf(t) + 1.0f) * (1.0f / 3.0f);  // /log2(8)
  return (t < 0.f) ? -v : v;
}

__global__ void k_cpb(const float* __restrict__ w1, const float* __restrict__ b1,
                      const float* __restrict__ w2, float* __restrict__ tab) {
  int tid = blockIdx.x * blockDim.x + threadIdx.x;
  if (tid >= 225) return;
  int i = tid / 15, j = tid % 15;
  float vh = rctv(i), vw = rctv(j);
  float acc[8] = {0.f, 0.f, 0.f, 0.f, 0.f, 0.f, 0.f, 0.f};
  for (int jj = 0; jj < 512; ++jj) {
    float hs = vh * w1[jj] + vw * w1[512 + jj] + b1[jj];
    if (hs > 0.f) {
#pragma unroll
      for (int h = 0; h < 8; ++h) acc[h] += hs * w2[jj * 8 + h];
    }
  }
#pragma unroll
  for (int h = 0; h < 8; ++h) tab[tid * 8 + h] = acc[h];
}

// K1b: bias16[h][i][j] = 16*sigmoid(tab[rpi(i,j)][h])
__global__ void k_bias16(const float* __restrict__ tab,
                         float* __restrict__ bias16) {
  int i = blockIdx.x, j = threadIdx.x;
  int rel = ((i >> 3) - (j >> 3) + 7) * 15 + ((i & 7) - (j & 7) + 7);
#pragma unroll
  for (int h = 0; h < 8; ++h) {
    float t = tab[rel * 8 + h];
    bias16[h * 4096 + i * 64 + j] = 16.f / (1.f + expf(-t));
  }
}

// ---------------------------------------------------------------------------
// K3: shifted-window gather + QKV GEMM + bias + l2norm(q,k)
// grid 4096 (win, half-window of 32 tokens), block 256 (8 waves)
// LDS: xt 32x256 f16 (16KB) + ob 32x768 f32 (96KB) = 112KB
// ---------------------------------------------------------------------------
__global__ __launch_bounds__(256) void k_qkv(const float* __restrict__ x,
    const h16* __restrict__ qkvT, const float* __restrict__ qkv_bias,
    h16* __restrict__ qkv_h) {
  extern __shared__ char smem[];
  h16* xt = (h16*)smem;                       // 32 x 256
  float* ob = (float*)(smem + 32 * 256 * 2);  // 32 x 768
  int blk = blockIdx.x;
  int win = blk >> 1, half = blk & 1;
  int b = win >> 6, widx = win & 63;
  int wh = widx >> 3, ww = widx & 7;
  int r0 = half * 32;
  int tid = threadIdx.x;
  // gather x with cyclic shift (-4,-4), convert to f16
  for (int idx = tid; idx < 32 * 64; idx += 256) {
    int i = idx >> 6, c4 = idx & 63;
    int t = r0 + i;
    int r = t >> 3, c = t & 7;
    int h = (wh * 8 + r + 4) & 63;
    int w = (ww * 8 + c + 4) & 63;
    float4 v = ((const float4*)x)[(size_t)(b * 4096 + h * 64 + w) * 64 + c4];
    h16* d = xt + i * 256 + c4 * 4;
    d[0] = (h16)v.x; d[1] = (h16)v.y; d[2] = (h16)v.z; d[3] = (h16)v.w;
  }
  __syncthreads();
  int wv = tid >> 5;
  // each wave owns 6 nt columns; B fragments loaded once, reused for 2 M tiles
#pragma unroll 1
  for (int i = 0; i < 6; ++i) {
    int nt = wv + 8 * i;
    v8f a0 = {0.f, 0.f, 0.f, 0.f, 0.f, 0.f, 0.f, 0.f};
    v8f a1 = a0;
#pragma unroll
    for (int ks = 0; ks < 8; ++ks) {
      v16h bf = ldfragBT(qkvT + nt * 16 * 256 + ks * 32, 256);
      v16h f0 = ldfragA(xt + ks * 32, 256);
      v16h f1 = ldfragA(xt + 16 * 256 + ks * 32, 256);
      a0 = wmma16(f0, bf, a0);
      a1 = wmma16(f1, bf, a1);
    }
    float bv = qkv_bias[nt * 16 + (tid & 15)];
#pragma unroll
    for (int r = 0; r < 8; ++r) { a0[r] += bv; a1[r] += bv; }
    stD(ob + nt * 16, 768, a0);
    stD(ob + 16 * 768 + nt * 16, 768, a1);
  }
  __syncthreads();
  // normalize q,k rows (per head), relayout to [win][part][head][tok][d]
  for (int rt = tid; rt < 768; rt += 256) {
    int part = rt >> 8;  // 0=q,1=k,2=v
    int rem = rt & 255;
    int tok = rem >> 3, head = rem & 7;
    const float* src = ob + tok * 768 + part * 256 + head * 32;
    float rs = 1.f;
    if (part < 2) {
      float s = 0.f;
#pragma unroll
      for (int d = 0; d < 32; ++d) s += src[d] * src[d];
      rs = rsqrtf(fmaxf(s, 1e-12f));
    }
    h16* dst = qkv_h +
        (((size_t)(win * 3 + part) * 8 + head) * 64 + (r0 + tok)) * 32;
#pragma unroll
    for (int d = 0; d < 32; ++d) dst[d] = (h16)(src[d] * rs);
  }
}

// ---------------------------------------------------------------------------
// K4: cosine attention per (window, head). block 128 (4 waves)
// LDS: q 4KB, k 4KB, vT 4KB, S 16KB f32, P 8KB f16 = 36KB
// ---------------------------------------------------------------------------
__global__ __launch_bounds__(128) void k_attn(const h16* __restrict__ qkv_h,
    const float* __restrict__ logit_scale, const float* __restrict__ bias16,
    h16* __restrict__ aw_h) {
  extern __shared__ char smem[];
  h16* qs = (h16*)smem;              // 64x32 @ 0
  h16* ksd = qs + 2048;              // 64x32 @ 4096
  h16* vt = ksd + 2048;              // 32x64 @ 8192 (v transposed)
  float* sb = (float*)(vt + 2048);   // 64x64 f32 @ 12288
  h16* pb = (h16*)(sb + 4096);       // 64x64 f16 @ 28672
  int win = blockIdx.x, head = blockIdx.y;
  int tid = threadIdx.x;
  const h16* qg = qkv_h + ((size_t)win * 24 + head) * 2048;
  const h16* kg = qg + 8 * 2048;
  const h16* vg = qg + 16 * 2048;
  // async contiguous copies of q,k into LDS
  u64 qa = (u64)(size_t)qg, ka = (u64)(size_t)kg;
  for (int idx = tid; idx < 256; idx += 128) {
    async_cp16((u32)(idx * 16), qa + (u64)idx * 16);
    async_cp16((u32)(4096 + idx * 16), ka + (u64)idx * 16);
  }
  // transpose v into LDS (element scatter, keep on VALU path)
  for (int idx = tid; idx < 2048; idx += 128) {
    int t = idx >> 5, d = idx & 31;
    vt[d * 64 + t] = vg[idx];
  }
  wait_async0();
  __syncthreads();
  int wv = tid >> 5;
  {  // S = q @ k^T  (contraction over d=32)
    v16h a = ldfragA(qs + wv * 16 * 32, 32);
#pragma unroll
    for (int nt = 0; nt < 4; ++nt) {
      v16h bf = ldfragBT(ksd + nt * 16 * 32, 32);
      v8f acc = {0.f, 0.f, 0.f, 0.f, 0.f, 0.f, 0.f, 0.f};
      acc = wmma16(a, bf, acc);
      stD(sb + wv * 16 * 64 + nt * 16, 64, acc);
    }
  }
  __syncthreads();
  float scale = expf(fminf(logit_scale[head], 4.605170185988091f));
  {  // softmax: 2 lanes per row
    int row = tid >> 1, sub = tid & 1;
    int widx = win & 63;
    int regA = regionOf(widx, row);
    const float* brow = bias16 + head * 4096 + row * 64;
    float* srow = sb + row * 64 + sub * 32;
    float mx = -3.4e38f;
    for (int c2 = 0; c2 < 32; ++c2) {
      int col = sub * 32 + c2;
      float v = srow[c2] * scale + brow[col];
      if (regionOf(widx, col) != regA) v -= 100.f;
      srow[c2] = v;
      mx = fmaxf(mx, v);
    }
    mx = fmaxf(mx, __shfl_xor(mx, 1));
    float sum = 0.f;
    for (int c2 = 0; c2 < 32; ++c2) {
      float e = expf(srow[c2] - mx);
      sum += e;
      srow[c2] = e;
    }
    sum += __shfl_xor(sum, 1);
    float inv = 1.f / sum;
    h16* prow = pb + row * 64 + sub * 32;
    for (int c2 = 0; c2 < 32; ++c2) prow[c2] = (h16)(srow[c2] * inv);
  }
  __syncthreads();
#pragma unroll
  for (int nt = 0; nt < 2; ++nt) {  // out = P @ V (16x32 per wave)
    v8f acc = {0.f, 0.f, 0.f, 0.f, 0.f, 0.f, 0.f, 0.f};
#pragma unroll
    for (int kk = 0; kk < 2; ++kk) {
      v16h a = ldfragA(pb + wv * 16 * 64 + kk * 32, 64);
      v16h bf = ldfragBT(vt + nt * 16 * 64 + kk * 32, 64);
      acc = wmma16(a, bf, acc);
    }
    int lane = tid & 31, n = lane & 15, hi = lane >> 4;
#pragma unroll
    for (int r = 0; r < 8; ++r) {
      int m = wv * 16 + r + 8 * hi;
      aw_h[((size_t)win * 64 + m) * 256 + head * 32 + nt * 16 + n] =
          (h16)acc[r];
    }
  }
}

// ---------------------------------------------------------------------------
// K5: proj GEMM + window reverse + unshift + residual + LayerNorm1
// grid 4096 (win, half), block 256.  LDS: at 16KB + ob 32KB = 48KB
// ---------------------------------------------------------------------------
__global__ __launch_bounds__(256) void k_proj(const h16* __restrict__ aw_h,
    const h16* __restrict__ projT, const float* __restrict__ proj_b,
    const float* __restrict__ x, const float* __restrict__ g1,
    const float* __restrict__ bb1, float* __restrict__ x1f,
    h16* __restrict__ x1h) {
  extern __shared__ char smem[];
  h16* at = (h16*)smem;                // 32x256 @ 0
  float* ob = (float*)(smem + 16384);  // 32x256
  int blk = blockIdx.x;
  int win = blk >> 1, half = blk & 1;
  int tid = threadIdx.x;
  const h16* src = aw_h + ((size_t)win * 64 + half * 32) * 256;
  u64 sa = (u64)(size_t)src;
  for (int idx = tid; idx < 1024; idx += 256)
    async_cp16((u32)(idx * 16), sa + (u64)idx * 16);
  wait_async0();
  __syncthreads();
  int wv = tid >> 5;
#pragma unroll 1
  for (int i = 0; i < 2; ++i) {  // each wave: 2 nt columns x 2 M tiles
    int nt = wv + 8 * i;
    v8f a0 = {0.f, 0.f, 0.f, 0.f, 0.f, 0.f, 0.f, 0.f};
    v8f a1 = a0;
#pragma unroll
    for (int ks = 0; ks < 8; ++ks) {
      v16h bf = ldfragBT(projT + nt * 16 * 256 + ks * 32, 256);
      v16h f0 = ldfragA(at + ks * 32, 256);
      v16h f1 = ldfragA(at + 16 * 256 + ks * 32, 256);
      a0 = wmma16(f0, bf, a0);
      a1 = wmma16(f1, bf, a1);
    }
    float pbv = proj_b[nt * 16 + (tid & 15)];
#pragma unroll
    for (int r = 0; r < 8; ++r) { a0[r] += pbv; a1[r] += pbv; }
    stD(ob + nt * 16, 256, a0);
    stD(ob + 16 * 256 + nt * 16, 256, a1);
  }
  __syncthreads();
  // LN + residual: 8 threads per token row
  int row = tid >> 3, sub = tid & 7;
  float s = 0.f, sq = 0.f;
  for (int j = 0; j < 32; ++j) {
    float v = ob[row * 256 + sub * 32 + j];
    s += v;
    sq += v * v;
  }
  s += __shfl_xor(s, 1); sq += __shfl_xor(sq, 1);
  s += __shfl_xor(s, 2); sq += __shfl_xor(sq, 2);
  s += __shfl_xor(s, 4); sq += __shfl_xor(sq, 4);
  float mu = s * (1.f / 256.f);
  float var = sq * (1.f / 256.f) - mu * mu;
  float rs = rsqrtf(var + 1e-5f);
  int b = win >> 6, widx = win & 63;
  int wh = widx >> 3, ww = widx & 7;
  int tok = half * 32 + row;
  int r = tok >> 3, c = tok & 7;
  int h = (wh * 8 + r + 4) & 63;
  int w = (ww * 8 + c + 4) & 63;
  size_t tk = (size_t)b * 4096 + h * 64 + w;
  for (int j = 0; j < 32; ++j) {
    int ch = sub * 32 + j;
    float y = (ob[row * 256 + ch] - mu) * rs * g1[ch] + bb1[ch];
    float o = x[tk * 256 + ch] + y;
    x1f[tk * 256 + ch] = o;
    x1h[tk * 256 + ch] = (h16)o;
  }
}

// ---------------------------------------------------------------------------
// K6: fused MLP (fc1 + exact GELU + fc2) + LayerNorm2 + residual -> out
// grid 4096 (32 tokens each), block 256.  LDS: xt 16KB + hb 64KB = 80KB
// ---------------------------------------------------------------------------
__global__ __launch_bounds__(256) void k_mlp(const h16* __restrict__ x1h,
    const float* __restrict__ x1f, const h16* __restrict__ fc1T,
    const float* __restrict__ fc1_b, const h16* __restrict__ fc2T,
    const float* __restrict__ fc2_b, const float* __restrict__ g2,
    const float* __restrict__ bb2, float* __restrict__ out) {
  extern __shared__ char smem[];
  h16* xt = (h16*)smem;            // 32x256 f16 @ 0
  h16* hb = (h16*)(smem + 16384);  // 32x1024 f16; reused as 32x256 f32
  int blk = blockIdx.x, tid = threadIdx.x;
  size_t t0 = (size_t)blk * 32;
  const h16* src = x1h + t0 * 256;
  u64 sa = (u64)(size_t)src;
  for (int idx = tid; idx < 1024; idx += 256)
    async_cp16((u32)(idx * 16), sa + (u64)idx * 16);
  wait_async0();
  __syncthreads();
  int wv = tid >> 5;
  int lane = tid & 31, n16 = lane & 15, hi = lane >> 4;
  // fc1 + gelu: each wave 8 nt columns x 2 M tiles
#pragma unroll 1
  for (int i = 0; i < 8; ++i) {
    int nt = wv + 8 * i;
    v8f a0 = {0.f, 0.f, 0.f, 0.f, 0.f, 0.f, 0.f, 0.f};
    v8f a1 = a0;
#pragma unroll
    for (int ks = 0; ks < 8; ++ks) {
      v16h bf = ldfragBT(fc1T + nt * 16 * 256 + ks * 32, 256);
      v16h f0 = ldfragA(xt + ks * 32, 256);
      v16h f1 = ldfragA(xt + 16 * 256 + ks * 32, 256);
      a0 = wmma16(f0, bf, a0);
      a1 = wmma16(f1, bf, a1);
    }
    float bv = fc1_b[nt * 16 + n16];
#pragma unroll
    for (int r = 0; r < 8; ++r) {
      float v0 = a0[r] + bv, v1 = a1[r] + bv;
      v0 = 0.5f * v0 * (1.f + erff(v0 * 0.70710678f));
      v1 = 0.5f * v1 * (1.f + erff(v1 * 0.70710678f));
      hb[(r + 8 * hi) * 1024 + nt * 16 + n16] = (h16)v0;
      hb[(16 + r + 8 * hi) * 1024 + nt * 16 + n16] = (h16)v1;
    }
  }
  __syncthreads();
  // fc2: each wave 2 nt columns x 2 M tiles, accumulate in registers
  v8f acc2[2][2];
#pragma unroll 1
  for (int i = 0; i < 2; ++i) {
    int nt = wv + 8 * i;
    v8f a0 = {0.f, 0.f, 0.f, 0.f, 0.f, 0.f, 0.f, 0.f};
    v8f a1 = a0;
#pragma unroll 4
    for (int ks = 0; ks < 32; ++ks) {
      v16h bf = ldfragBT(fc2T + nt * 16 * 1024 + ks * 32, 1024);
      v16h f0 = ldfragA(hb + ks * 32, 1024);
      v16h f1 = ldfragA(hb + 16 * 1024 + ks * 32, 1024);
      a0 = wmma16(f0, bf, a0);
      a1 = wmma16(f1, bf, a1);
    }
    float bv = fc2_b[nt * 16 + n16];
#pragma unroll
    for (int r = 0; r < 8; ++r) { a0[r] += bv; a1[r] += bv; }
    acc2[i][0] = a0;
    acc2[i][1] = a1;
  }
  __syncthreads();  // all reads of hb complete
  float* ob = (float*)hb;
#pragma unroll
  for (int i = 0; i < 2; ++i) {
    int nt = wv + 8 * i;
    stD(ob + nt * 16, 256, acc2[i][0]);
    stD(ob + 16 * 256 + nt * 16, 256, acc2[i][1]);
  }
  __syncthreads();
  // LN2 + residual
  int row = tid >> 3, sub = tid & 7;
  float s = 0.f, sq = 0.f;
  for (int j = 0; j < 32; ++j) {
    float v = ob[row * 256 + sub * 32 + j];
    s += v;
    sq += v * v;
  }
  s += __shfl_xor(s, 1); sq += __shfl_xor(sq, 1);
  s += __shfl_xor(s, 2); sq += __shfl_xor(sq, 2);
  s += __shfl_xor(s, 4); sq += __shfl_xor(sq, 4);
  float mu = s * (1.f / 256.f);
  float var = sq * (1.f / 256.f) - mu * mu;
  float rsv = rsqrtf(var + 1e-5f);
  size_t tk = t0 + row;
  for (int j = 0; j < 32; ++j) {
    int ch = sub * 32 + j;
    float y = (ob[row * 256 + ch] - mu) * rsv * g2[ch] + bb2[ch];
    out[tk * 256 + ch] = x1f[tk * 256 + ch] + y;
  }
}

// ---------------------------------------------------------------------------
extern "C" void kernel_launch(void* const* d_in, const int* in_sizes, int n_in,
                              void* d_out, int out_size, void* d_ws,
                              size_t ws_size, hipStream_t stream) {
  (void)in_sizes; (void)n_in; (void)out_size; (void)ws_size;
  const float* x           = (const float*)d_in[0];
  const float* qkv_w       = (const float*)d_in[1];
  const float* q_bias      = (const float*)d_in[2];
  const float* v_bias      = (const float*)d_in[3];
  const float* logit_scale = (const float*)d_in[4];
  const float* cpb_w1      = (const float*)d_in[5];
  const float* cpb_b1      = (const float*)d_in[6];
  const float* cpb_w2      = (const float*)d_in[7];
  const float* proj_w      = (const float*)d_in[8];
  const float* proj_b      = (const float*)d_in[9];
  const float* n1g         = (const float*)d_in[10];
  const float* n1b         = (const float*)d_in[11];
  const float* fc1_w       = (const float*)d_in[12];
  const float* fc1_b       = (const float*)d_in[13];
  const float* fc2_w       = (const float*)d_in[14];
  const float* fc2_b       = (const float*)d_in[15];
  const float* n2g         = (const float*)d_in[16];
  const float* n2b         = (const float*)d_in[17];

  char* ws = (char*)d_ws;
  size_t off = 0;
  auto take = [&](size_t bytes) -> char* {
    char* p = ws + off;
    off = (off + bytes + 255) & ~(size_t)255;
    return p;
  };
  h16* qkvT      = (h16*)take(768 * 256 * 2);
  h16* projT     = (h16*)take(256 * 256 * 2);
  h16* fc1T      = (h16*)take(1024 * 256 * 2);
  h16* fc2T      = (h16*)take(256 * 1024 * 2);
  float* qkvBias = (float*)take(768 * 4);
  float* cpbTab  = (float*)take(225 * 8 * 4);
  float* bias16  = (float*)take(8 * 64 * 64 * 4);
  char* qkvRegion = take((size_t)2048 * 3 * 8 * 64 * 32 * 2);  // 201 MB
  h16* qkv_h = (h16*)qkvRegion;
  h16* aw_h  = (h16*)take((size_t)2048 * 64 * 256 * 2);        // 67 MB
  // x1 reuses the qkv region (dead after attention): 134MB f32 + 67MB f16
  float* x1f = (float*)qkvRegion;
  h16* x1h   = (h16*)(qkvRegion + (size_t)134217728);

  // K0: weight prep
  k_transpose_h16<<<(768 * 256 + 255) / 256, 256, 0, stream>>>(qkv_w, qkvT, 256, 768);
  k_transpose_h16<<<(256 * 256 + 255) / 256, 256, 0, stream>>>(proj_w, projT, 256, 256);
  k_transpose_h16<<<(1024 * 256 + 255) / 256, 256, 0, stream>>>(fc1_w, fc1T, 256, 1024);
  k_transpose_h16<<<(256 * 1024 + 255) / 256, 256, 0, stream>>>(fc2_w, fc2T, 1024, 256);
  k_qkvbias<<<3, 256, 0, stream>>>(q_bias, v_bias, qkvBias);
  // K1: relative-position bias
  k_cpb<<<1, 256, 0, stream>>>(cpb_w1, cpb_b1, cpb_w2, cpbTab);
  k_bias16<<<64, 64, 0, stream>>>(cpbTab, bias16);
  // K3: QKV
  k_qkv<<<4096, 256, 114688, stream>>>(x, qkvT, qkvBias, qkv_h);
  // K4: attention
  k_attn<<<dim3(2048, 8), 128, 36864, stream>>>(qkv_h, logit_scale, bias16, aw_h);
  // K5: proj + reverse/unshift + residual + LN1
  k_proj<<<4096, 256, 49152, stream>>>(aw_h, projT, proj_b, x, n1g, n1b, x1f, x1h);
  // K6: MLP + LN2 + residual
  k_mlp<<<4096, 256, 81920, stream>>>(x1h, x1f, fc1T, fc1_b, fc2T, fc2_b, n2g,
                                      n2b, (float*)d_out);
}
